// TransformerDecoder_50165218018178
// MI455X (gfx1250) — matrix operand order repaired and checked by reading, not compiled, not run on hardware
//
#include <hip/hip_runtime.h>

// ---------------------------------------------------------------------------
// CDNA5 / gfx1250 transformer decoder forward.
// Matmuls: v_wmma_f32_16x16x32_bf16 (wave32), fp32 accumulate.
// Weights pre-converted to bf16 + pre-transposed to [N][K]; activations flow
// as bf16 row-major (converted once at production). GEMM inner loop is a
// 2-stage software pipeline: double-buffered LDS fed by
// global_load_async_to_lds_b128 (ASYNCcnt), overlapped with the WMMA block.
// ---------------------------------------------------------------------------

typedef __attribute__((ext_vector_type(16))) __bf16 v16bf;
typedef __attribute__((ext_vector_type(8)))  __bf16 v8bf;
typedef __attribute__((ext_vector_type(8)))  float  v8f;

// round-to-nearest-even fp32 -> bf16
__device__ __forceinline__ __bf16 f2bf(float f) {
  union { float f; unsigned u; } in; in.f = f;
  unsigned r = in.u + 0x7FFFu + ((in.u >> 16) & 1u);
  unsigned short h = (unsigned short)(r >> 16);
  __bf16 b;
  __builtin_memcpy(&b, &h, 2);
  return b;
}

__device__ __forceinline__ v8f wmma_bf16(v16bf a, v16bf b, v8f c) {
  return __builtin_amdgcn_wmma_f32_16x16x32_bf16(false, a, false, b, (short)0, c,
                                                 false, false);
}

// Async DMA: 16 bytes global -> LDS, tracked by ASYNCcnt.
__device__ __forceinline__ void async_g2l_b128(const void* lds_ptr,
                                               const void* gptr) {
  unsigned loff = (unsigned)(size_t)lds_ptr;   // low 32b = LDS aperture offset
  asm volatile("global_load_async_to_lds_b128 %0, %1, off"
               :: "v"(loff), "v"(gptr) : "memory");
}
__device__ __forceinline__ void wait_async0() {
  asm volatile("s_wait_asynccnt 0" ::: "memory");
}
__device__ __forceinline__ void wait_async4() {   // allow newest stage in flight
  asm volatile("s_wait_asynccnt 4" ::: "memory");
}

// A-matrix fragment (16xK32 bf16) from LDS, row-major, leading dim ldk.
__device__ __forceinline__ v16bf frag_a_lds(const __bf16* base, int ldk) {
  const int lane = threadIdx.x & 31;
  const int m  = lane & 15;
  const int k0 = (lane & 16) ? 8 : 0;
  v8bf lo = *(const v8bf*)(base + m * ldk + k0);
  v8bf hi = *(const v8bf*)(base + m * ldk + k0 + 16);
  return __builtin_shufflevector(lo, hi, 0,1,2,3,4,5,6,7,8,9,10,11,12,13,14,15);
}

// B-matrix fragment (K32x16) from LDS stored TRANSPOSED as Bt[N][K].
__device__ __forceinline__ v16bf frag_b_lds(const __bf16* baseT, int ldk) {
  const int lane = threadIdx.x & 31;
  const int n  = lane & 15;
  const int k0 = (lane & 16) ? 8 : 0;
  v8bf lo = *(const v8bf*)(baseT + n * ldk + k0);
  v8bf hi = *(const v8bf*)(baseT + n * ldk + k0 + 16);
  return __builtin_shufflevector(lo, hi, 0,1,2,3,4,5,6,7,8,9,10,11,12,13,14,15);
}

// A fragment straight from a global bf16 matrix (row stride ld), k offset koff.
__device__ __forceinline__ v16bf frag_a_global_bf16(const __bf16* rowbase,
                                                    int ld, int koff) {
  const int lane = threadIdx.x & 31;
  const int m  = lane & 15;
  const int k0 = koff + ((lane & 16) ? 8 : 0);
  const __bf16* p = rowbase + (size_t)m * ld;
  v8bf lo = *(const v8bf*)(p + k0);
  v8bf hi = *(const v8bf*)(p + k0 + 16);
  return __builtin_shufflevector(lo, hi, 0,1,2,3,4,5,6,7,8,9,10,11,12,13,14,15);
}

// ---------------------------------------------------------------------------
// Embedding: x = tok_emb[idx] + pos_emb  (fp32 residual chain + bf16 mirror)
// ---------------------------------------------------------------------------
__global__ __launch_bounds__(256)
void embed_kernel(const int* __restrict__ idx, const float* __restrict__ tok,
                  const float* __restrict__ pos, float* __restrict__ x,
                  __bf16* __restrict__ xb, int BT, int T, int D) {
  int i = blockIdx.x * blockDim.x + threadIdx.x;
  if (i >= BT * D) return;
  int d  = i % D;
  int bt = i / D;
  int t  = bt % T;
  float v = tok[(size_t)idx[bt] * D + d] + pos[(size_t)t * D + d];
  x[i]  = v;
  xb[i] = f2bf(v);
}

// ---------------------------------------------------------------------------
// Weight prep: Wt[n][k] = bf16(W[k][n]); W is [K][N] row-major.
// grid (N/64, K/64, Z); per-z strides handle the head-blocked QKV layout.
// ---------------------------------------------------------------------------
__global__ __launch_bounds__(256)
void transpose_conv_kernel(const float* __restrict__ W, __bf16* __restrict__ Wt,
                           int K, int N, size_t in_zstride, size_t out_zstride) {
  __shared__ float Ts[64][65];
  const float* Wz  = W  + (size_t)blockIdx.z * in_zstride;
  __bf16*      Wtz = Wt + (size_t)blockIdx.z * out_zstride;
  const int nb = blockIdx.x * 64, kb = blockIdx.y * 64;
  const int tr = threadIdx.x >> 2;        // 0..63
  const int cc = (threadIdx.x & 3) * 16;  // 0,16,32,48

  const float4* src = (const float4*)(Wz + (size_t)(kb + tr) * N + nb + cc);
#pragma unroll
  for (int j = 0; j < 4; ++j) {
    float4 vv = src[j];
    Ts[tr][cc + 4 * j + 0] = vv.x;
    Ts[tr][cc + 4 * j + 1] = vv.y;
    Ts[tr][cc + 4 * j + 2] = vv.z;
    Ts[tr][cc + 4 * j + 3] = vv.w;
  }
  __syncthreads();
  __bf16* dst = Wtz + (size_t)(nb + tr) * K + kb + cc;
#pragma unroll
  for (int j = 0; j < 16; ++j) dst[j] = f2bf(Ts[cc + j][tr]);
}

// ---------------------------------------------------------------------------
// WMMA GEMM: out = A[M,K](bf16) @ Wt[N,K](bf16,transposed) (+bias/resid/relu)
// outf (fp32) and/or outbf (bf16) may each be null.
// Block tile 128x128, 256 threads = 8 waves (4x2); 8 WMMA per wave per K-step.
// 2-stage pipeline: async DMA for stage s+1 runs under stage-s WMMAs.
// ---------------------------------------------------------------------------
__global__ __launch_bounds__(256)
void gemm_wmma_kernel(const __bf16* __restrict__ Abf,
                      const __bf16* __restrict__ Wt,
                      const float* __restrict__ bias,
                      const float* __restrict__ resid,
                      float* __restrict__ outf, __bf16* __restrict__ outbf,
                      int M, int N, int K, int relu) {
  __shared__ alignas(32) __bf16 As[2][128 * 32];   // 128 m x 32 k, x2 stages
  __shared__ alignas(32) __bf16 Bs[2][128 * 32];   // 128 n x 32 k ([N][K])

  const int t    = threadIdx.x;
  const int wid  = t >> 5;
  const int wm   = wid >> 1;        // 0..3 -> rows wm*32
  const int wn   = wid & 1;         // 0..1 -> cols wn*64
  const int lane = t & 31;
  const int mb   = blockIdx.y * 128;
  const int nb   = blockIdx.x * 128;

  v8f acc[2][4];
#pragma unroll
  for (int mt = 0; mt < 2; ++mt)
#pragma unroll
    for (int nt = 0; nt < 4; ++nt) acc[mt][nt] = (v8f){};

  const int row = t >> 1;           // 0..127 (m for A, n for B)
  const int kc  = (t & 1) * 16;     // 0,16

  const __bf16* asrc = Abf + (size_t)(mb + row) * K + kc;
  const __bf16* bsrc = Wt  + (size_t)(nb + row) * K + kc;

  const int nsteps = K >> 5;

  auto issue = [&](int s) {
    const int k0 = s << 5;
    __bf16* ad = As[s & 1] + row * 32 + kc;
    __bf16* bd = Bs[s & 1] + row * 32 + kc;
    async_g2l_b128(ad,     asrc + k0);
    async_g2l_b128(ad + 8, asrc + k0 + 8);
    async_g2l_b128(bd,     bsrc + k0);
    async_g2l_b128(bd + 8, bsrc + k0 + 8);
  };

  issue(0);
  for (int s = 0; s < nsteps; ++s) {
    if (s + 1 < nsteps) {
      issue(s + 1);      // stage s+1 DMA flies under stage-s WMMAs
      wait_async4();     // retire stage s (per-wave async loads are in-order)
    } else {
      wait_async0();     // drain on the last stage
    }
    __syncthreads();     // stage-s tiles visible block-wide

    const __bf16* Ab = As[s & 1];
    const __bf16* Bb = Bs[s & 1];
    v16bf af[2];
#pragma unroll
    for (int mt = 0; mt < 2; ++mt)
      af[mt] = frag_a_lds(Ab + (wm * 32 + mt * 16) * 32, 32);
#pragma unroll
    for (int nt = 0; nt < 4; ++nt) {
      v16bf bf = frag_b_lds(Bb + (wn * 64 + nt * 16) * 32, 32);
#pragma unroll
      for (int mt = 0; mt < 2; ++mt)
        acc[mt][nt] = wmma_bf16(af[mt], bf, acc[mt][nt]);
    }
    __syncthreads();     // protect buffer (s&1) before stage s+2 overwrites it
  }

  // --- epilogue ---
  const int hi8   = (lane & 16) ? 8 : 0;
  const int nlane = lane & 15;
  float bv[4];
#pragma unroll
  for (int nt = 0; nt < 4; ++nt) {
    const int gn = nb + wn * 64 + nt * 16 + nlane;
    bv[nt] = bias ? bias[gn] : 0.0f;
  }
#pragma unroll
  for (int mt = 0; mt < 2; ++mt) {
#pragma unroll
    for (int r = 0; r < 8; ++r) {
      const int gr = mb + wm * 32 + mt * 16 + r + hi8;
#pragma unroll
      for (int nt = 0; nt < 4; ++nt) {
        const int gn = nb + wn * 64 + nt * 16 + nlane;
        float vv = acc[mt][nt][r] + bv[nt];
        if (resid) vv += resid[(size_t)gr * N + gn];
        if (relu) vv = vv > 0.0f ? vv : 0.0f;
        if (outf)  outf[(size_t)gr * N + gn]  = vv;
        if (outbf) outbf[(size_t)gr * N + gn] = f2bf(vv);
      }
    }
  }
}

// ---------------------------------------------------------------------------
// Flash attention (causal), bf16 q/k/v in (B,T,D), head h at column h*64.
// Block = (qtile of 128 rows, h, b); 8 waves x 16 query rows; HS = 64.
// Output written bf16 (all the proj GEMM consumes).
// ---------------------------------------------------------------------------
__global__ __launch_bounds__(256)
void attention_kernel(const __bf16* __restrict__ q, const __bf16* __restrict__ k,
                      const __bf16* __restrict__ v, __bf16* __restrict__ out,
                      int T, int D, float scale) {
  __shared__ alignas(32) __bf16 Ks[64 * 64];        // [s][c]  (Bt for q.kT)
  __shared__ alignas(32) __bf16 Vs[64 * 64];        // [c][s]  (Bt for p.v)
  __shared__ alignas(32) __bf16 Pl[8 * 16 * 64];    // per-wave P tiles

  const int t    = threadIdx.x;
  const int wid  = t >> 5;
  const int lane = t & 31;
  const int qt = blockIdx.x, h = blockIdx.y, b = blockIdx.z;

  const int hi8   = (lane & 16) ? 8 : 0;
  const int nlane = lane & 15;
  const int qr0   = qt * 128 + wid * 16;

  const __bf16* qrow = q + ((size_t)(b * T + qr0)) * D + h * 64;
  v16bf qf0 = frag_a_global_bf16(qrow, D, 0);
  v16bf qf1 = frag_a_global_bf16(qrow, D, 32);

  v8f   o[4];
  float mrow[8], lrow[8];
#pragma unroll
  for (int i = 0; i < 4; ++i) o[i] = (v8f){};
#pragma unroll
  for (int r = 0; r < 8; ++r) { mrow[r] = -1e30f; lrow[r] = 0.0f; }

  const int srow = t >> 2;          // 0..63
  const int c0   = (t & 3) * 16;    // 0,16,32,48
  const int nst  = 2 * (qt + 1);    // causal upper bound, uniform per block

  __bf16* myP  = Pl + wid * (16 * 64);
  __bf16* kdst = Ks + srow * 64 + c0;

  for (int st = 0; st < nst; ++st) {
    const int s0 = st * 64;
    __syncthreads();
    {
      const __bf16* kp = k + ((size_t)(b * T + s0 + srow)) * D + h * 64 + c0;
      const __bf16* vp = v + ((size_t)(b * T + s0 + srow)) * D + h * 64 + c0;
      // K tile: pure async DMA (row-major copy)
      async_g2l_b128(kdst,     kp);
      async_g2l_b128(kdst + 8, kp + 8);
      // V tile: transposed scatter (raw bf16, no conversion)
      v8bf v0 = *(const v8bf*)vp;
      v8bf v1 = *(const v8bf*)(vp + 8);
#pragma unroll
      for (int j = 0; j < 8; ++j) {
        Vs[(c0 + j) * 64 + srow]     = v0[j];
        Vs[(c0 + 8 + j) * 64 + srow] = v1[j];
      }
    }
    wait_async0();
    __syncthreads();

    v8f S[4];
#pragma unroll
    for (int nt = 0; nt < 4; ++nt) {
      S[nt] = (v8f){};
      S[nt] = wmma_bf16(qf0, frag_b_lds(Ks + (nt * 16) * 64, 64), S[nt]);
      S[nt] = wmma_bf16(qf1, frag_b_lds(Ks + (nt * 16) * 64 + 32, 64), S[nt]);
    }

    float tmax[8];
#pragma unroll
    for (int r = 0; r < 8; ++r) tmax[r] = -1e30f;
#pragma unroll
    for (int nt = 0; nt < 4; ++nt) {
      const int sc = s0 + nt * 16 + nlane;
#pragma unroll
      for (int r = 0; r < 8; ++r) {
        const int qr = qr0 + r + hi8;
        float sv = S[nt][r] * scale;
        sv = (sc <= qr) ? sv : -1e30f;
        S[nt][r] = sv;
        tmax[r] = fmaxf(tmax[r], sv);
      }
    }
#pragma unroll
    for (int mk = 1; mk <= 8; mk <<= 1)
#pragma unroll
      for (int r = 0; r < 8; ++r) tmax[r] = fmaxf(tmax[r], __shfl_xor(tmax[r], mk));

    float alpha[8];
#pragma unroll
    for (int r = 0; r < 8; ++r) {
      float mn = fmaxf(mrow[r], tmax[r]);
      alpha[r] = __expf(mrow[r] - mn);
      mrow[r]  = mn;
      lrow[r] *= alpha[r];
    }
#pragma unroll
    for (int nt = 0; nt < 4; ++nt)
#pragma unroll
      for (int r = 0; r < 8; ++r) o[nt][r] *= alpha[r];

    float rsum[8];
#pragma unroll
    for (int r = 0; r < 8; ++r) rsum[r] = 0.0f;
#pragma unroll
    for (int nt = 0; nt < 4; ++nt) {
#pragma unroll
      for (int r = 0; r < 8; ++r) {
        float p = __expf(S[nt][r] - mrow[r]);
        rsum[r] += p;
        myP[(r + hi8) * 64 + nt * 16 + nlane] = f2bf(p);
      }
    }
#pragma unroll
    for (int mk = 1; mk <= 8; mk <<= 1)
#pragma unroll
      for (int r = 0; r < 8; ++r) rsum[r] += __shfl_xor(rsum[r], mk);
#pragma unroll
    for (int r = 0; r < 8; ++r) lrow[r] += rsum[r];

#pragma unroll
    for (int j = 0; j < 2; ++j) {
      v16bf pf = frag_a_lds(myP + j * 32, 64);
#pragma unroll
      for (int nt = 0; nt < 4; ++nt) {
        v16bf vf = frag_b_lds(Vs + (nt * 16) * 64 + j * 32, 64);
        o[nt] = wmma_bf16(pf, vf, o[nt]);
      }
    }
  }

#pragma unroll
  for (int r = 0; r < 8; ++r) {
    const int qr = qr0 + r + hi8;
    const float inv = 1.0f / lrow[r];
    __bf16* op = out + ((size_t)(b * T + qr)) * D + h * 64;
#pragma unroll
    for (int nt = 0; nt < 4; ++nt) op[nt * 16 + nlane] = f2bf(o[nt][r] * inv);
  }
}

// ---------------------------------------------------------------------------
// Host-side orchestration (graph-capture safe: kernel launches only)
// ---------------------------------------------------------------------------
extern "C" void kernel_launch(void* const* d_in, const int* in_sizes, int n_in,
                              void* d_out, int out_size, void* d_ws,
                              size_t ws_size, hipStream_t stream) {
  const int*   idx   = (const int*)d_in[0];
  const float* tok   = (const float*)d_in[1];
  const float* pos   = (const float*)d_in[2];
  const float* Wq    = (const float*)d_in[3];
  const float* Wk    = (const float*)d_in[4];
  const float* Wv    = (const float*)d_in[5];
  const float* Wproj = (const float*)d_in[6];
  const float* bproj = (const float*)d_in[7];
  const float* W1    = (const float*)d_in[8];
  const float* b1    = (const float*)d_in[9];
  const float* W2    = (const float*)d_in[10];
  const float* b2    = (const float*)d_in[11];
  const float* Wlm   = (const float*)d_in[12];
  const float* blm   = (const float*)d_in[13];
  float* out = (float*)d_out;

  const int B = 2, T = 1024, D = 1024, H = 16, L = 4, V = 32000, HS = 64;
  const int F = 4 * D;
  const int M = B * T;
  const float scale = 0.03125f;  // D^-0.5 (reference scales by D, not HS)

  // --- workspace carve ---
  float* ws = (float*)d_ws;
  float* x  = ws;                                 // fp32 residual chain, M x D
  __bf16* bfb = (__bf16*)(x + (size_t)M * D);
  __bf16* xb  = bfb;                              // bf16 mirror of x
  __bf16* qbb = xb  + (size_t)M * D;
  __bf16* kbb = qbb + (size_t)M * D;
  __bf16* vbb = kbb + (size_t)M * D;
  __bf16* atb = vbb + (size_t)M * D;
  __bf16* hbb = atb + (size_t)M * D;              // M x F
  __bf16* wtb = hbb + (size_t)M * F;
  const size_t szDD = (size_t)D * D;
  const size_t szDF = (size_t)D * F;
  __bf16* wq_t = wtb;                             // L x [D][D]  (n = h*64+c)
  __bf16* wk_t = wq_t + (size_t)L * szDD;
  __bf16* wv_t = wk_t + (size_t)L * szDD;
  __bf16* wp_t = wv_t + (size_t)L * szDD;         // L x [D][D]
  __bf16* w1_t = wp_t + (size_t)L * szDD;         // L x [F][D]
  __bf16* w2_t = w1_t + (size_t)L * szDF;         // L x [D][F]
  __bf16* wl_t = w2_t + (size_t)L * szDF;         // [V][D]

  embed_kernel<<<(M * D + 255) / 256, 256, 0, stream>>>(idx, tok, pos, x, xb,
                                                        M, T, D);

  const dim3 blk(256);

  // --- weight prep (once per launch) ---
  for (int l = 0; l < L; ++l) {
    const size_t wqkv = (size_t)l * H * D * HS;
    transpose_conv_kernel<<<dim3(1, D / 64, H), blk, 0, stream>>>(
        Wq + wqkv, wq_t + (size_t)l * szDD, D, HS, (size_t)D * HS, (size_t)HS * D);
    transpose_conv_kernel<<<dim3(1, D / 64, H), blk, 0, stream>>>(
        Wk + wqkv, wk_t + (size_t)l * szDD, D, HS, (size_t)D * HS, (size_t)HS * D);
    transpose_conv_kernel<<<dim3(1, D / 64, H), blk, 0, stream>>>(
        Wv + wqkv, wv_t + (size_t)l * szDD, D, HS, (size_t)D * HS, (size_t)HS * D);
    transpose_conv_kernel<<<dim3(D / 64, D / 64), blk, 0, stream>>>(
        Wproj + (size_t)l * szDD, wp_t + (size_t)l * szDD, D, D, 0, 0);
    transpose_conv_kernel<<<dim3(F / 64, D / 64), blk, 0, stream>>>(
        W1 + (size_t)l * szDF, w1_t + (size_t)l * szDF, D, F, 0, 0);
    transpose_conv_kernel<<<dim3(D / 64, F / 64), blk, 0, stream>>>(
        W2 + (size_t)l * szDF, w2_t + (size_t)l * szDF, F, D, 0, 0);
  }
  transpose_conv_kernel<<<dim3(V / 64, D / 64), blk, 0, stream>>>(
      Wlm, wl_t, D, V, 0, 0);

  const dim3 gD(D / 128, M / 128);
  const dim3 gF(F / 128, M / 128);
  const dim3 gV(V / 128, M / 128);
  const dim3 gA(T / 128, H, B);

  for (int l = 0; l < L; ++l) {
    gemm_wmma_kernel<<<gD, blk, 0, stream>>>(xb, wq_t + (size_t)l * szDD,
                                             nullptr, nullptr, nullptr, qbb,
                                             M, D, D, 0);
    gemm_wmma_kernel<<<gD, blk, 0, stream>>>(xb, wk_t + (size_t)l * szDD,
                                             nullptr, nullptr, nullptr, kbb,
                                             M, D, D, 0);
    gemm_wmma_kernel<<<gD, blk, 0, stream>>>(xb, wv_t + (size_t)l * szDD,
                                             nullptr, nullptr, nullptr, vbb,
                                             M, D, D, 0);
    attention_kernel<<<gA, blk, 0, stream>>>(qbb, kbb, vbb, atb, T, D, scale);
    gemm_wmma_kernel<<<gD, blk, 0, stream>>>(atb, wp_t + (size_t)l * szDD,
                                             bproj + (size_t)l * D, x, x, xb,
                                             M, D, D, 0);
    gemm_wmma_kernel<<<gF, blk, 0, stream>>>(xb, w1_t + (size_t)l * szDF,
                                             b1 + (size_t)l * F, nullptr,
                                             nullptr, hbb, M, F, D, 1);
    gemm_wmma_kernel<<<gD, blk, 0, stream>>>(hbb, w2_t + (size_t)l * szDF,
                                             b2 + (size_t)l * D, x, x, xb,
                                             M, D, F, 0);
  }
  gemm_wmma_kernel<<<gV, blk, 0, stream>>>(xb, wl_t, blm, nullptr, out, nullptr,
                                           M, V, D, 0);
}
